// TemporalMultiHeadAttention_52587579572483
// MI455X (gfx1250) — matrix operand order repaired and checked by reading, not compiled, not run on hardware
//
#include <hip/hip_runtime.h>
#include <stdint.h>

// ---------------------------------------------------------------------------
// MHA forward for gfx1250 (MI455X).
//  * All matrix math via v_wmma_f32_16x16x32_bf16 (wave32).
//  * GEMM staging via global_load_async_to_lds_b128 + s_wait_asynccnt,
//    double-buffered LDS (CDNA5 async-copy pipeline).
// ---------------------------------------------------------------------------

typedef __bf16 bf16;
typedef __attribute__((ext_vector_type(16))) __bf16 v16bf;
typedef __attribute__((ext_vector_type(8)))  __bf16 bf16x8;
typedef __attribute__((ext_vector_type(4)))  __bf16 bf16x4;
typedef __attribute__((ext_vector_type(8)))  float  v8f;
typedef __attribute__((ext_vector_type(4)))  float  f32x4;

#define D_MODEL 1024
#define SEQ     2048
#define BATCH   4
#define HEADS   16
#define HDIM    64
#define NTOK    (BATCH * SEQ)   // 8192

static __device__ __forceinline__ v8f wmma_bf16(v16bf a, v16bf b, v8f c) {
  // (neg_a, A, neg_b, B, c_mod, C, reuse_a, reuse_b)
  return __builtin_amdgcn_wmma_f32_16x16x32_bf16(false, a, false, b, (short)0, c,
                                                 false, false);
}

// A operand (16x32 bf16, M per lane): element e -> k = half*16 + e
// => two contiguous 8xbf16 (16B) chunks at k = half*16 and k = half*16 + 8.
static __device__ __forceinline__ v16bf fragA(const bf16* row, int half) {
  bf16x8 lo = *(const bf16x8*)(row + half * 16);
  bf16x8 hi = *(const bf16x8*)(row + half * 16 + 8);
  v16bf r;
#pragma unroll
  for (int i = 0; i < 8; ++i) { r[i] = lo[i]; r[i + 8] = hi[i]; }
  return r;
}
// B operand (32x16 bf16, N per lane): element e -> k = half*16 + e.
static __device__ __forceinline__ v16bf fragB(const bf16* row, int half) {
  bf16x8 lo = *(const bf16x8*)(row + half * 16);
  bf16x8 hi = *(const bf16x8*)(row + half * 16 + 8);
  v16bf r;
#pragma unroll
  for (int i = 0; i < 8; ++i) { r[i] = lo[i]; r[i + 8] = hi[i]; }
  return r;
}

// Async 16B global->LDS copy (ASYNCcnt-tracked). LDS byte address is the low
// 32 bits of the generic pointer (LDS aperture keeps offset in addr[31:0]).
static __device__ __forceinline__ void stage16(const bf16* g, bf16* l) {
  asm volatile("global_load_async_to_lds_b128 %0, %1, off"
               :: "v"((unsigned)(uintptr_t)l), "v"(g) : "memory");
}

// ---------------------------------------------------------------------------
// Kernel 1a: transpose + convert the four weight matrices to bf16, Wt[out][in].
// ---------------------------------------------------------------------------
__global__ __launch_bounds__(256) void wtrans_kernel(
    const float* __restrict__ W0, const float* __restrict__ W1,
    const float* __restrict__ W2, const float* __restrict__ W3,
    bf16* __restrict__ WtBase) {
  const float* W = (blockIdx.z == 0) ? W0 : (blockIdx.z == 1) ? W1
                 : (blockIdx.z == 2) ? W2 : W3;
  bf16* Wt = WtBase + (size_t)blockIdx.z * D_MODEL * D_MODEL;
  __shared__ float tile[32][33];
  const int x  = blockIdx.x * 32 + threadIdx.x;   // out column n
  const int y0 = blockIdx.y * 32 + threadIdx.y;   // in row k
#pragma unroll
  for (int i = 0; i < 4; ++i)
    tile[threadIdx.y + i * 8][threadIdx.x] = W[(size_t)(y0 + i * 8) * D_MODEL + x];
  __syncthreads();
  const int xo = blockIdx.y * 32 + threadIdx.x;   // k
  const int yo = blockIdx.x * 32 + threadIdx.y;   // n
#pragma unroll
  for (int i = 0; i < 4; ++i)
    Wt[(size_t)(yo + i * 8) * D_MODEL + xo] = (bf16)tile[threadIdx.x][threadIdx.y + i * 8];
}

// ---------------------------------------------------------------------------
// Kernel 1b: convert activations x (f32) -> xh (bf16) so GEMM staging can use
// async LDS copies (no conversion possible in the DMA path).
// ---------------------------------------------------------------------------
__global__ __launch_bounds__(256) void xcvt_kernel(const float* __restrict__ X,
                                                   bf16* __restrict__ Xh) {
  const size_t i = ((size_t)blockIdx.x * 256 + threadIdx.x) * 8;
  const f32x4 a = *(const f32x4*)&X[i];
  const f32x4 b = *(const f32x4*)&X[i + 4];
  bf16x8 p;
#pragma unroll
  for (int j = 0; j < 4; ++j) { p[j] = (bf16)a[j]; p[j + 4] = (bf16)b[j]; }
  *(bf16x8*)&Xh[i] = p;
}

// ---------------------------------------------------------------------------
// Kernel 2/4: WMMA GEMM  Y = X @ W + b, X bf16 [8192,1024].
// Block: 256 threads = 8 waves; block tile 128x128; wave tile 32x64
// (8 WMMAs per 32-deep K-step). Double-buffered LDS filled with
// global_load_async_to_lds_b128; s_wait_asynccnt pipelining.
//  mode 0: Q -> [B,H,S,Dh] bf16, pre-scaled by Dh^-1/2
//  mode 1: K -> [B,H,S,Dh] bf16
//  mode 2: V -> [B,H,Dh,S] bf16 (transposed for the PV WMMA B-operand)
//  mode 3: out -> f32 [NTOK,1024] + bias
// ---------------------------------------------------------------------------
__global__ __launch_bounds__(256) void gemm_kernel(
    const bf16* __restrict__ X, const bf16* __restrict__ WtBase,
    const float* __restrict__ bq, const float* __restrict__ bk,
    const float* __restrict__ bv, const float* __restrict__ bo,
    bf16* __restrict__ dstQ, bf16* __restrict__ dstK, bf16* __restrict__ dstV,
    float* __restrict__ outF, int modeBase) {
  const int mode = modeBase + (int)blockIdx.z;
  const int tid  = threadIdx.x;
  const int lane = tid & 31, wave = tid >> 5;
  const int half = lane >> 4, l15 = lane & 15;
  const int row0 = blockIdx.x * 128;   // token rows
  const int col0 = blockIdx.y * 128;   // output columns
  const bf16*  Wt   = WtBase + (size_t)mode * D_MODEL * D_MODEL;
  const float* bias = (mode == 0) ? bq : (mode == 1) ? bk : (mode == 2) ? bv : bo;

  __shared__ bf16 Xs[2][128 * 40];   // 128x32 bf16, stride 40 (16B chunks)
  __shared__ bf16 Ws[2][128 * 40];   // 128 output cols x 32 K

  const int wr = (wave >> 1) * 32;   // wave rows (4 waves in M)
  const int wn = (wave & 1) * 64;    // wave cols (2 waves in N)

  const v8f vzero = {0.f, 0.f, 0.f, 0.f, 0.f, 0.f, 0.f, 0.f};
  v8f acc[2][4];
#pragma unroll
  for (int mt = 0; mt < 2; ++mt)
#pragma unroll
    for (int nt = 0; nt < 4; ++nt) acc[mt][nt] = vzero;

  // Stager: thread -> (row sr, 16B chunk sc); two rows per array per step.
  const int sr = tid >> 2;          // 0..63
  const int sc = (tid & 3) * 8;     // 0,8,16,24
  const bf16* xg = X  + (size_t)(row0 + sr) * D_MODEL + sc;
  const bf16* wg = Wt + (size_t)(col0 + sr) * D_MODEL + sc;

  auto issue = [&](int buf, int kk) {
    stage16(xg + kk,                &Xs[buf][sr * 40 + sc]);
    stage16(xg + kk + 64 * D_MODEL, &Xs[buf][(sr + 64) * 40 + sc]);
    stage16(wg + kk,                &Ws[buf][sr * 40 + sc]);
    stage16(wg + kk + 64 * D_MODEL, &Ws[buf][(sr + 64) * 40 + sc]);
  };

  issue(0, 0);
  for (int s = 0; s < D_MODEL / 32; ++s) {
    if (s + 1 < D_MODEL / 32) {
      issue((s + 1) & 1, (s + 1) * 32);
      asm volatile("s_wait_asynccnt 0x4" ::: "memory");  // current step done
    } else {
      asm volatile("s_wait_asynccnt 0x0" ::: "memory");
    }
    __syncthreads();  // all waves' async writes for this buffer visible

    const bf16* xs  = Xs[s & 1];
    const bf16* wsb = Ws[s & 1];
    v16bf a[2], b[4];
#pragma unroll
    for (int t = 0; t < 2; ++t)
      a[t] = fragA(&xs[(wr + t * 16 + l15) * 40], half);
#pragma unroll
    for (int n = 0; n < 4; ++n)
      b[n] = fragB(&wsb[(wn + n * 16 + l15) * 40], half);
#pragma unroll
    for (int mt = 0; mt < 2; ++mt)
#pragma unroll
      for (int nt = 0; nt < 4; ++nt)
        acc[mt][nt] = wmma_bf16(a[mt], b[nt], acc[mt][nt]);
    __syncthreads();  // done reading before buffer is re-filled in s+1
  }

  // Epilogue. C/D layout: lane column N = l15, VGPR v -> row M = v + half*8.
#pragma unroll
  for (int mt = 0; mt < 2; ++mt)
#pragma unroll
    for (int nt = 0; nt < 4; ++nt) {
      const int colg   = col0 + wn + nt * 16 + l15;
      const float bval = bias[colg];
      const int rbase  = row0 + wr + mt * 16 + half * 8;
      if (mode == 3) {
#pragma unroll
        for (int v = 0; v < 8; ++v)
          outF[(size_t)(rbase + v) * D_MODEL + colg] = acc[mt][nt][v] + bval;
      } else if (mode == 2) {  // V, transposed layout [B,H,Dh,S]
        const int b_ = rbase >> 11, s0 = rbase & (SEQ - 1);
        const int h_ = colg >> 6, d_ = colg & 63;
        bf16x8 pk;
#pragma unroll
        for (int v = 0; v < 8; ++v) pk[v] = (bf16)(acc[mt][nt][v] + bval);
        *(bf16x8*)&dstV[(((size_t)b_ * HEADS + h_) * HDIM + d_) * SEQ + s0] = pk;
      } else {                 // Q (pre-scaled) or K, layout [B,H,S,Dh]
        bf16* dst = (mode == 0) ? dstQ : dstK;
        const float sc_ = (mode == 0) ? 0.125f : 1.0f;  // 64^-0.5
        const int h_ = colg >> 6, d_ = colg & 63;
#pragma unroll
        for (int v = 0; v < 8; ++v) {
          const int t_ = rbase + v;
          const int b_ = t_ >> 11, s_ = t_ & (SEQ - 1);
          dst[(((size_t)b_ * HEADS + h_) * SEQ + s_) * HDIM + d_] =
              (bf16)((acc[mt][nt][v] + bval) * sc_);
        }
      }
    }
}

// ---------------------------------------------------------------------------
// Kernel 3: causal flash attention, score-transposed so P feeds the PV WMMA
// with no cross-lane movement. One wave = 16 queries, key blocks of 32.
//   S^T = K_A x Q^T_B   (contraction over Dh=64, two chained WMMAs per tile)
//   O^T = V^T_A x P^T_B (contraction over 32 keys)
// K A-fragment key rows are permuted (tile t, row M -> key t*8 + (M&8)*2 + (M&7))
// so the score C-layout rows land exactly where the P^T B-layout needs them.
// ---------------------------------------------------------------------------
__global__ __launch_bounds__(256) void attn_kernel(
    const bf16* __restrict__ Q, const bf16* __restrict__ K,
    const bf16* __restrict__ Vt, bf16* __restrict__ Oh) {
  const int tid  = threadIdx.x;
  const int lane = tid & 31;
  const int wave = tid >> 5;
  const int half = lane >> 4;
  const int l15  = lane & 15;
  const int bh   = blockIdx.x;                  // 0..63 = b*16+h
  const int qt   = blockIdx.y * 8 + wave;       // query tile 0..127
  const int qb   = qt * 16;
  const int bidx = bh >> 4, hidx = bh & 15;
  const bf16* Qp = Q  + (size_t)bh * SEQ * HDIM;
  const bf16* Kp = K  + (size_t)bh * SEQ * HDIM;
  const bf16* Vp = Vt + (size_t)bh * HDIM * SEQ;   // [Dh][S]
  const int q = qb + l15;                          // this lane's query row

  v16bf qf[2];                                     // Q^T B-fragments, d chunks
#pragma unroll
  for (int c = 0; c < 2; ++c)
    qf[c] = fragB(&Qp[(size_t)q * HDIM + c * 32], half);

  const v8f vzero = {0.f, 0.f, 0.f, 0.f, 0.f, 0.f, 0.f, 0.f};
  v8f o[4];                                        // O^T accum, d chunks of 16
#pragma unroll
  for (int c = 0; c < 4; ++c) o[c] = vzero;
  float m = -1e30f, l = 0.f;

  const int nblk = (qb + 47) >> 5;                 // ceil((qb+16)/32)
  for (int j = 0; j < nblk; ++j) {
    const int kb = j * 32;
    v8f st[2];
#pragma unroll
    for (int t = 0; t < 2; ++t) {
      const int keyrow = kb + t * 8 + ((l15 & 8) << 1) + (l15 & 7);
      v8f s = vzero;
#pragma unroll
      for (int c = 0; c < 2; ++c)
        s = wmma_bf16(fragA(&Kp[(size_t)keyrow * HDIM + c * 32], half), qf[c], s);
      st[t] = s;  // lane: query=l15; vgpr v holds key kb + t*8 + half*16 + v
    }
    if (kb + 31 > qb) {  // diagonal block: causal mask
#pragma unroll
      for (int t = 0; t < 2; ++t)
#pragma unroll
        for (int v = 0; v < 8; ++v) {
          const int key = kb + t * 8 + half * 16 + v;
          if (key > q) st[t][v] = -1e30f;
        }
    }
    // online softmax: this lane holds keys [kb+half*16, kb+half*16+16)
    float mx = st[0][0];
#pragma unroll
    for (int v = 1; v < 8; ++v) mx = fmaxf(mx, st[0][v]);
#pragma unroll
    for (int v = 0; v < 8; ++v) mx = fmaxf(mx, st[1][v]);
    mx = fmaxf(mx, __shfl_xor(mx, 16, 32));
    const float mn   = fmaxf(m, mx);
    const float corr = __expf(m - mn);
    float rs = 0.f;
    v16bf pf;  // P^T B-fragment: element e -> key kb + half*16 + e
#pragma unroll
    for (int v = 0; v < 8; ++v) {
      const float p0 = __expf(st[0][v] - mn);
      const float p1 = __expf(st[1][v] - mn);
      rs += p0 + p1;
      pf[v]     = (bf16)p0;
      pf[v + 8] = (bf16)p1;
    }
    rs += __shfl_xor(rs, 16, 32);
    l = l * corr + rs;
    m = mn;
#pragma unroll
    for (int c = 0; c < 4; ++c) {  // O^T += V^T x P^T  per 16-wide d chunk
      const v16bf va = fragA(&Vp[(size_t)(c * 16 + l15) * SEQ + kb], half);
#pragma unroll
      for (int v = 0; v < 8; ++v) o[c][v] *= corr;
      o[c] = wmma_bf16(va, pf, o[c]);
    }
  }

  const float invl = 1.f / l;
  // O^T D-layout: lane = query col l15, vgpr v -> d = c*16 + half*8 + v.
  const size_t rowbase = ((size_t)bidx * SEQ + q) * D_MODEL + hidx * HDIM;
#pragma unroll
  for (int c = 0; c < 4; ++c) {
    bf16x8 pk;
#pragma unroll
    for (int v = 0; v < 8; ++v) pk[v] = (bf16)(o[c][v] * invl);
    *(bf16x8*)&Oh[rowbase + c * 16 + half * 8] = pk;
  }
}

// ---------------------------------------------------------------------------
// Host entry. Workspace layout (bytes):
//   [0,8M)    Wt: 4 x 1024x1024 bf16 (Q,K,V,O transposed weights)
//   [8M,24M)  xh: activations bf16 [NTOK,1024]
//   [24M,40M) Q   [B,H,S,Dh] bf16 (pre-scaled by Dh^-1/2)
//   [40M,56M) K   [B,H,S,Dh] bf16
//   [56M,72M) Vt  [B,H,Dh,S] bf16
//   [72M,88M) Oh  [NTOK,1024] bf16 (attention output, token-major)
// ---------------------------------------------------------------------------
extern "C" void kernel_launch(void* const* d_in, const int* in_sizes, int n_in,
                              void* d_out, int out_size, void* d_ws, size_t ws_size,
                              hipStream_t stream) {
  (void)in_sizes; (void)n_in; (void)out_size; (void)ws_size;
  const float* x  = (const float*)d_in[0];
  const float* Wq = (const float*)d_in[1];
  const float* bq = (const float*)d_in[2];
  const float* Wk = (const float*)d_in[3];
  const float* bk = (const float*)d_in[4];
  const float* Wv = (const float*)d_in[5];
  const float* bv = (const float*)d_in[6];
  const float* Wo = (const float*)d_in[7];
  const float* bo = (const float*)d_in[8];
  float* out = (float*)d_out;

  char* ws = (char*)d_ws;
  const size_t MB = 1024 * 1024;
  bf16* Wt = (bf16*)(ws);
  bf16* xh = (bf16*)(ws + 8 * MB);
  bf16* Qb = (bf16*)(ws + 24 * MB);
  bf16* Kb = (bf16*)(ws + 40 * MB);
  bf16* Vb = (bf16*)(ws + 56 * MB);
  bf16* Oh = (bf16*)(ws + 72 * MB);

  wtrans_kernel<<<dim3(32, 32, 4), dim3(32, 8), 0, stream>>>(Wq, Wk, Wv, Wo, Wt);
  xcvt_kernel<<<dim3((NTOK * D_MODEL) / 2048), dim3(256), 0, stream>>>(x, xh);
  gemm_kernel<<<dim3(NTOK / 128, D_MODEL / 128, 3), dim3(256), 0, stream>>>(
      xh, Wt, bq, bk, bv, bo, Qb, Kb, Vb, nullptr, 0);
  attn_kernel<<<dim3(BATCH * HEADS, SEQ / (16 * 8)), dim3(256), 0, stream>>>(
      Qb, Kb, Vb, Oh);
  gemm_kernel<<<dim3(NTOK / 128, D_MODEL / 128, 1), dim3(256), 0, stream>>>(
      Oh, Wt, bq, bk, bv, bo, nullptr, nullptr, nullptr, out, 3);
}
